// CausalSelfAttention_13640816132460
// MI455X (gfx1250) — matrix operand order repaired and checked by reading, not compiled, not run on hardware
//
#include <hip/hip_runtime.h>

// ---------------------------------------------------------------------------
// Types
// ---------------------------------------------------------------------------
typedef __bf16 bf16;
typedef bf16 v16bf __attribute__((ext_vector_type(16)));
typedef bf16 v8bf  __attribute__((ext_vector_type(8)));
typedef float v8f  __attribute__((ext_vector_type(8)));

#define B_   2
#define T_   2048
#define C_   2048
#define H_   16
#define G_   4
#define HS_  128
#define SWS_ 1024
#define NQKV 3072   // (H + 2G) * HS

// gfx1250 has hardware f32->bf16 converts; let the compiler emit v_cvt_*bf16
static __device__ __forceinline__ bf16 f2bf(float f) { return (bf16)f; }

static __device__ __forceinline__ v16bf combine(v8bf a, v8bf b) {
    return __builtin_shufflevector(a, b, 0,1,2,3,4,5,6,7,8,9,10,11,12,13,14,15);
}

// ---------------------------------------------------------------------------
// Async global -> LDS copy (16 bytes / lane), ASYNCcnt-tracked on gfx1250.
// Builtin signature (per hipcc diagnostic): pointee is int vector_size(16),
// args (global v4i*, lds v4i*, imm offset, imm cpol).
// Low 32 bits of a generic LDS pointer are the LDS byte address on AMDGPU.
// ---------------------------------------------------------------------------
typedef int v4i __attribute__((__vector_size__(16)));
#if __has_builtin(__builtin_amdgcn_global_load_async_to_lds_b128)
#define HAVE_ASYNC_LDS 1
typedef __attribute__((address_space(1))) v4i* gv4i_p;
typedef __attribute__((address_space(3))) v4i* lv4i_p;
static __device__ __forceinline__ void async_cp16(bf16* lds, const bf16* g) {
    __builtin_amdgcn_global_load_async_to_lds_b128(
        (gv4i_p)(unsigned long long)g,
        (lv4i_p)(unsigned int)(unsigned long long)lds,
        0, 0);
}
#else
#define HAVE_ASYNC_LDS 0
static __device__ __forceinline__ void async_cp16(bf16* lds, const bf16* g) {
    *(v8bf*)lds = *(const v8bf*)g;   // sync fallback through VGPRs
}
#endif

#if __has_builtin(__builtin_amdgcn_s_wait_asynccnt)
#define WAIT_ASYNC(n) __builtin_amdgcn_s_wait_asynccnt(n)
#else
#define WAIT_ASYNC(n) asm volatile("s_wait_asynccnt %0" :: "i"(n) : "memory")
#endif

// ---------------------------------------------------------------------------
// fp32 -> bf16 conversion
// ---------------------------------------------------------------------------
__global__ void cvt_f32_bf16(const float* __restrict__ src, bf16* __restrict__ dst, int n) {
    int i = blockIdx.x * blockDim.x + threadIdx.x;
    int stride = gridDim.x * blockDim.x;
    for (; i < n; i += stride) dst[i] = f2bf(src[i]);
}

// ---------------------------------------------------------------------------
// C[M,Nc] = A[M,Kc](bf16) @ B[Nc,Kc]^T(bf16)   (both row-major, W stored [N,K])
// Block: 256 threads = 8 waves, tile 128(M) x 128(N); wave = 32x64 sub-tile
// (2x4 = 8 WMMAs per K-step against 2 A + 4 B fragment loads).
// A/B tiles double-buffered in LDS via async-to-LDS loads (4 async ops per
// wave per K-step -> s_wait_asynccnt 4 overlaps next-tile loads with compute).
// Nc/Kc are compile-time so the epilogue folds r*Nc into store immediates.
// ---------------------------------------------------------------------------
template<int Nc, int Kc>
__global__ void __launch_bounds__(256) gemm_bf16(const bf16* __restrict__ A,
                                                 const bf16* __restrict__ Bm,
                                                 float* __restrict__ Cout) {
    __shared__ bf16 As[2][128 * 32];
    __shared__ bf16 Bs[2][128 * 32];

    const int tid  = threadIdx.x;
    const int lane = tid & 31;
    const int wave = tid >> 5;
    const int half = lane >> 4;
    const int l16  = lane & 15;
    const int wm   = wave & 3;          // 4 M strips of 32
    const int wn   = wave >> 2;         // 2 N strips of 64
    const int mBase = blockIdx.x * 128;
    const int nBase = blockIdx.y * 128;

    // staging map: each 128x32 tile = 512 x 16B chunks -> 2 chunks/thread/tile
    const int arow = tid >> 2;          // 0..63 ; second chunk row = arow + 64
    const int acol = (tid & 3) * 8;

    const bf16* gA0 = A  + (size_t)(mBase + arow) * Kc + acol;
    const bf16* gA1 = A  + (size_t)(mBase + arow + 64) * Kc + acol;
    const bf16* gB0 = Bm + (size_t)(nBase + arow) * Kc + acol;
    const bf16* gB1 = Bm + (size_t)(nBase + arow + 64) * Kc + acol;

#define STAGE(k0, buf) do {                                        \
        async_cp16(&As[buf][arow * 32 + acol],        gA0 + (k0)); \
        async_cp16(&As[buf][(arow + 64) * 32 + acol], gA1 + (k0)); \
        async_cp16(&Bs[buf][arow * 32 + acol],        gB0 + (k0)); \
        async_cp16(&Bs[buf][(arow + 64) * 32 + acol], gB1 + (k0)); \
    } while (0)

    STAGE(0, 0);
    v8f acc[2][4] = {};

    for (int k0 = 0; k0 < Kc; k0 += 32) {
        const int buf = (k0 >> 5) & 1;
        if (k0 + 32 < Kc) {
            STAGE(k0 + 32, buf ^ 1);
            WAIT_ASYNC(4);           // current tile's 4 ops done; next 4 in flight
        } else {
            WAIT_ASYNC(0);
        }
        __syncthreads();             // all waves' portions of this tile visible

        v16bf a[2], b[4];
#pragma unroll
        for (int tm = 0; tm < 2; ++tm) {
            const bf16* p = &As[buf][(wm * 32 + tm * 16 + l16) * 32 + half * 16];
            a[tm] = combine(*(const v8bf*)p, *(const v8bf*)(p + 8));
        }
#pragma unroll
        for (int tn = 0; tn < 4; ++tn) {
            b[tn] = *(const v16bf*)&Bs[buf][(wn * 64 + tn * 16 + l16) * 32 + half * 16];
        }
#pragma unroll
        for (int tm = 0; tm < 2; ++tm)
#pragma unroll
            for (int tn = 0; tn < 4; ++tn)
                acc[tm][tn] = __builtin_amdgcn_wmma_f32_16x16x32_bf16(
                    false, a[tm], false, b[tn], (short)0, acc[tm][tn], false, false);

        __syncthreads();             // everyone done reading before buf is reloaded
    }
#undef STAGE

    // Epilogue: one base pointer, all offsets compile-time -> immediate offsets
    float* crow = Cout + (size_t)(mBase + wm * 32 + 8 * half) * Nc + nBase + wn * 64 + l16;
#pragma unroll
    for (int tm = 0; tm < 2; ++tm)
#pragma unroll
        for (int tn = 0; tn < 4; ++tn)
#pragma unroll
            for (int r = 0; r < 8; ++r)
                crow[(size_t)(tm * 16 + r) * Nc + tn * 16] = acc[tm][tn][r];
}

// ---------------------------------------------------------------------------
// RoPE + repack:  qkv fp32 [B,T,G,6,HS] ->
//   Qb bf16 [B,H,T,HS], Kb bf16 [B,G,T,HS], Vt bf16 [B,G,HS,T] (V transposed)
// ---------------------------------------------------------------------------
__global__ void rope_pack(const float* __restrict__ qkv,
                          const float* __restrict__ cosT,
                          const float* __restrict__ sinT,
                          bf16* __restrict__ Qb,
                          bf16* __restrict__ Kb,
                          bf16* __restrict__ Vt) {
    const int blk = blockIdx.x;            // ((b*T)+t)*G + g
    const int g = blk & (G_ - 1);
    const int t = (blk >> 2) & (T_ - 1);
    const int b = blk >> 13;               // / (T_*G_)
    const int d = threadIdx.x;             // 0..127

    const float cs = cosT[t * HS_ + d];
    const float sn = sinT[t * HS_ + d];
    const float* base = qkv + (size_t)(b * T_ + t) * NQKV + g * (6 * HS_);

#pragma unroll
    for (int slot = 0; slot < 6; ++slot) {
        float v = base[slot * HS_ + d];
        float out;
        if (slot < 5) {  // slots 0..3 = q heads, slot 4 = k : apply RoPE
            float rot = (d < 64) ? -base[slot * HS_ + d + 64]
                                 :  base[slot * HS_ + d - 64];
            out = v * cs + rot * sn;
        } else {
            out = v;
        }
        bf16 ob = f2bf(out);
        if (slot < 4)
            Qb[((size_t)(b * H_ + g * 4 + slot) * T_ + t) * HS_ + d] = ob;
        else if (slot == 4)
            Kb[((size_t)(b * G_ + g) * T_ + t) * HS_ + d] = ob;
        else
            Vt[((size_t)(b * G_ + g) * HS_ + d) * T_ + t] = ob;
    }
}

// ---------------------------------------------------------------------------
// Sliding-window flash attention. One wave = one 16-query tile of one head.
// Block = 128 threads = 4 independent waves.
// ---------------------------------------------------------------------------
__global__ void __launch_bounds__(128) attn_swa(const bf16* __restrict__ Qb,
                                                const bf16* __restrict__ Kb,
                                                const bf16* __restrict__ Vt,
                                                bf16* __restrict__ Yb) {
    __shared__ bf16 Plds[4][16 * 32];      // per-wave 16x32 probability tile

    const int lane = threadIdx.x & 31;
    const int wave = threadIdx.x >> 5;
    const int half = lane >> 4;
    const int l16  = lane & 15;

    const int wid = blockIdx.x * 4 + wave;
    const int it  = wid & 127;             // T/16 tiles
    const int h   = (wid >> 7) & 15;
    const int b   = wid >> 11;
    const int g   = h >> 2;
    const int i0  = it * 16;

    const bf16* qbase = Qb + (size_t)(b * H_ + h) * T_ * HS_;
    const bf16* kbase = Kb + (size_t)(b * G_ + g) * T_ * HS_;
    const bf16* vbase = Vt + (size_t)(b * G_ + g) * HS_ * T_;

    // Q fragments (resident): 4 K-steps over HS=128
    v16bf qf[4];
#pragma unroll
    for (int kk = 0; kk < 4; ++kk) {
        const bf16* p = qbase + (size_t)(i0 + l16) * HS_ + kk * 32 + half * 16;
        qf[kk] = combine(*(const v8bf*)p, *(const v8bf*)(p + 8));
    }

    float m[8], l[8];
    v8f o[8] = {};
#pragma unroll
    for (int r = 0; r < 8; ++r) { m[r] = -1e30f; l[r] = 0.0f; }

    const float scale = 0.08838834764831845f;   // 1/sqrt(128)
    int s_lo = i0 + 15 - (SWS_ - 1);
    if (s_lo < 0) s_lo = 0;
    const int s_start = s_lo & ~31;

    for (int s0 = s_start; s0 <= i0 + 15; s0 += 32) {
        // ---- scores = Q @ K^T for keys [s0, s0+31] (always within [0,T)) ----
        v8f sc[2] = {};
#pragma unroll
        for (int sub = 0; sub < 2; ++sub) {
            const int key = s0 + sub * 16 + l16;
#pragma unroll
            for (int kk = 0; kk < 4; ++kk) {
                v16bf kf = *(const v16bf*)(kbase + (size_t)key * HS_ + kk * 32 + half * 16);
                sc[sub] = __builtin_amdgcn_wmma_f32_16x16x32_bf16(
                    false, qf[kk], false, kf, (short)0, sc[sub], false, false);
            }
        }

        // ---- mask + online softmax (rows 0-7 in lanes 0-15, rows 8-15 in 16-31) ----
#pragma unroll
        for (int r = 0; r < 8; ++r) {
            const int i = i0 + r + 8 * half;
#pragma unroll
            for (int sub = 0; sub < 2; ++sub) {
                const int j = s0 + sub * 16 + l16;
                const bool ok = (j <= i) && (j > i - SWS_);
                sc[sub][r] = ok ? sc[sub][r] * scale : -1e30f;
            }
            float v = fmaxf(sc[0][r], sc[1][r]);
            v = fmaxf(v, __shfl_xor(v, 8));
            v = fmaxf(v, __shfl_xor(v, 4));
            v = fmaxf(v, __shfl_xor(v, 2));
            v = fmaxf(v, __shfl_xor(v, 1));

            const float mn    = fmaxf(m[r], v);
            const float alpha = __expf(m[r] - mn);
            m[r] = mn;

            const float p0 = __expf(sc[0][r] - mn);
            const float p1 = __expf(sc[1][r] - mn);
            const int row = r + 8 * half;
            Plds[wave][row * 32 + l16]      = f2bf(p0);
            Plds[wave][row * 32 + 16 + l16] = f2bf(p1);

            float ps = p0 + p1;
            ps += __shfl_xor(ps, 8);
            ps += __shfl_xor(ps, 4);
            ps += __shfl_xor(ps, 2);
            ps += __shfl_xor(ps, 1);
            l[r] = l[r] * alpha + ps;

#pragma unroll
            for (int dt = 0; dt < 8; ++dt) o[dt][r] *= alpha;
        }

        // ---- O += P @ V ;  P A-fragment re-read from LDS in A-layout ----
        v8bf pc0 = *(const v8bf*)&Plds[wave][l16 * 32 + half * 16];
        v8bf pc1 = *(const v8bf*)&Plds[wave][l16 * 32 + half * 16 + 8];
        v16bf pf = combine(pc0, pc1);
#pragma unroll
        for (int dt = 0; dt < 8; ++dt) {
            v16bf vf = *(const v16bf*)(vbase + (size_t)(dt * 16 + l16) * T_ + s0 + half * 16);
            o[dt] = __builtin_amdgcn_wmma_f32_16x16x32_bf16(
                false, pf, false, vf, (short)0, o[dt], false, false);
        }
    }

    // ---- normalize + store bf16 to Yb [B,T,H*HS] ----
#pragma unroll
    for (int r = 0; r < 8; ++r) {
        const float inv = 1.0f / l[r];
        const int t = i0 + r + 8 * half;
        bf16* yrow = Yb + (size_t)(b * T_ + t) * (H_ * HS_) + h * HS_;
#pragma unroll
        for (int dt = 0; dt < 8; ++dt)
            yrow[dt * 16 + l16] = f2bf(o[dt][r] * inv);
    }
}

// ---------------------------------------------------------------------------
// Launch
// ---------------------------------------------------------------------------
extern "C" void kernel_launch(void* const* d_in, const int* in_sizes, int n_in,
                              void* d_out, int out_size, void* d_ws, size_t ws_size,
                              hipStream_t stream) {
    (void)in_sizes; (void)n_in; (void)out_size; (void)ws_size;

    const float* x    = (const float*)d_in[0];
    const float* cosT = (const float*)d_in[1];
    const float* sinT = (const float*)d_in[2];
    const float* Wa   = (const float*)d_in[3];
    const float* Wp   = (const float*)d_in[4];

    char* ws = (char*)d_ws;
    bf16*  Xb   = (bf16*)(ws);                    // 4096*2048*2  = 16,777,216
    bf16*  Wab  = (bf16*)(ws + 16777216);         // 3072*2048*2  = 12,582,912
    bf16*  Wpb  = (bf16*)(ws + 29360128);         // 2048*2048*2  =  8,388,608
    float* QKVf = (float*)(ws + 37748736);        // 4096*3072*4  = 50,331,648
    bf16*  Qb   = (bf16*)(ws + 88080384);         // 2*16*2048*128*2
    bf16*  Kb   = (bf16*)(ws + 104857600);        // 2*4*2048*128*2
    bf16*  Vt   = (bf16*)(ws + 109051904);        // 2*4*128*2048*2
    bf16*  Yb   = (bf16*)(ws + 113246208);        // 4096*2048*2

    cvt_f32_bf16<<<2048, 256, 0, stream>>>(x,  Xb,  B_ * T_ * C_);
    cvt_f32_bf16<<<2048, 256, 0, stream>>>(Wa, Wab, NQKV * C_);
    cvt_f32_bf16<<<2048, 256, 0, stream>>>(Wp, Wpb, C_ * H_ * HS_);

    // QKV projection: [4096 x 2048] @ [2048 x 3072]
    gemm_bf16<NQKV, C_><<<dim3((B_ * T_) / 128, NQKV / 128), 256, 0, stream>>>(Xb, Wab, QKVf);

    rope_pack<<<B_ * T_ * G_, HS_, 0, stream>>>(QKVf, cosT, sinT, Qb, Kb, Vt);

    attn_swa<<<(B_ * H_ * (T_ / 16)) / 4, 128, 0, stream>>>(Qb, Kb, Vt, Yb);

    // Output projection: [4096 x 2048] @ [2048 x 2048] -> fp32 d_out
    gemm_bf16<C_, H_ * HS_><<<dim3((B_ * T_) / 128, C_ / 128), 256, 0, stream>>>(
        Yb, Wpb, (float*)d_out);
}
